// NeuralTensorNetwork_36472862277763
// MI455X (gfx1250) — matrix operand order repaired and checked by reading, not compiled, not run on hardware
//
#include <hip/hip_runtime.h>
#include <hip/hip_bf16.h>
#include <cmath>

typedef __attribute__((ext_vector_type(16))) _Float16 v16h;
typedef __attribute__((ext_vector_type(8)))  _Float16 v8h;
typedef __attribute__((ext_vector_type(4)))  _Float16 v4h;
typedef __attribute__((ext_vector_type(8)))  float    v8f;

#define VOCAB 100000
#define DDIM  512
#define KDIM  256
#define BATCH 4096
#define NPH   8

union AFrag { v16h v; v8h g[2]; _Float16 h[16]; };
union CFrag { v8f  v; float f[8]; };

// K-index map for the 16-bit A fragment (16x32, MxK):
// h[0..7]  -> K = 8*hi + 0..7
// h[8..15] -> K = 16 + 8*hi + 0..7
__device__ __forceinline__ int a_kmap(int e, int hi) {
    int p = e >> 1, q = e & 1;
    return ((p >> 2) << 4) + ((p & 3) << 1) + q + (hi << 3);
}

// ---------------------------------------------------------------------------
// Kernel 1: phrase embedding  out[b,d] = sum_n w[b,n] * emb[ids[b,n], d]
// ---------------------------------------------------------------------------
__global__ void phrase_embed_kernel(const int* __restrict__ ids,
                                    const float* __restrict__ w,
                                    const float* __restrict__ emb,
                                    float* __restrict__ out) {
    const int b = blockIdx.x;
    int   id_[NPH];
    float wv_[NPH];
#pragma unroll
    for (int n = 0; n < NPH; ++n) {
        id_[n] = ids[b * NPH + n];
        wv_[n] = w[b * NPH + n];
    }
    for (int d = threadIdx.x; d < DDIM; d += blockDim.x) {
        float s = 0.f;
#pragma unroll
        for (int n = 0; n < NPH; ++n)
            s = fmaf(wv_[n], emb[(size_t)id_[n] * DDIM + d], s);
        out[(size_t)b * DDIM + d] = s;
    }
}

// ---------------------------------------------------------------------------
// Kernel 2: ACC[b,k] = bias[k] + concat(X,Y)[b,:] @ W[k,:]   (W is (K, 2*DIN))
// One wave per 16x16 output tile, pure-register WMMA GEMM.
// ---------------------------------------------------------------------------
__global__ __launch_bounds__(256)
void linear_wmma_kernel(const float* __restrict__ X,
                        const float* __restrict__ Y,
                        const float* __restrict__ W,
                        const float* __restrict__ bias,
                        float* __restrict__ ACC,
                        int DIN) {
    const int tid  = threadIdx.x;
    const int lane = tid & 31;
    const int wave = tid >> 5;
    const int hi   = lane >> 4;
    const int nn   = lane & 15;

    const int waveId = blockIdx.x * 8 + wave;
    const int ktiles = KDIM / 16;
    const int bt = waveId / ktiles;
    const int kt = waveId % ktiles;
    const int b0 = bt * 16, k0 = kt * 16;

    CFrag C;
#pragma unroll
    for (int r = 0; r < 8; ++r) C.f[r] = 0.f;

    const int twoD = 2 * DIN;
    for (int i = 0; i < twoD; i += 32) {
        AFrag A, Bf;
        const float* src;
        int coff;
        if (i < DIN) { src = X; coff = i; }
        else         { src = Y; coff = i - DIN; }
        const float* xrow = src + (size_t)(b0 + nn) * DIN + coff;
#pragma unroll
        for (int e = 0; e < 16; ++e)
            A.h[e] = (_Float16)xrow[16 * hi + e];
        const float* wrow = W + (size_t)(k0 + nn) * twoD + i + 16 * hi;
#pragma unroll
        for (int e = 0; e < 16; ++e)
            Bf.h[e] = (_Float16)wrow[e];
        C.v = __builtin_amdgcn_wmma_f32_16x16x32_f16(false, A.v, false, Bf.v,
                                                     (short)0, C.v, false, false);
    }
    const float bk = bias[k0 + nn];
#pragma unroll
    for (int r = 0; r < 8; ++r) {
        const int m = r + 8 * hi;
        ACC[(size_t)(b0 + m) * KDIM + k0 + nn] = C.f[r] + bk;
    }
}

// ---------------------------------------------------------------------------
// Kernel 3: OUT[b,k] = tanh( ACCin[b,k] + sum_{i,j} T[k,i,j]*X[b,i]*Y[b,j] )
// Grid: (B/128, K). Per WG: k fixed, 128 batch rows; T[k] streamed from
// global exactly once. X panel row-major f16 in LDS (b128 A-frag loads);
// T chunk stored TRANSPOSED (bshT[j][i]) so B-frags are contiguous b128 loads.
// ---------------------------------------------------------------------------
template<int DIN>
__global__ __launch_bounds__(256)
void bilinear_tanh_kernel(const float* __restrict__ T,
                          const float* __restrict__ X,
                          const float* __restrict__ Y,
                          const float* __restrict__ ACCin,
                          float* __restrict__ OUT) {
    constexpr int BT  = 128;     // batch rows per workgroup
    constexpr int IP  = 128;     // i-panel width
    constexpr int JC  = 64;      // j-chunk width
    constexpr int STR = IP + 24; // 304B row stride: 16B aligned, bank-rotating

    const int k    = blockIdx.y;
    const int b0   = blockIdx.x * BT;
    const int tid  = threadIdx.x;
    const int lane = tid & 31;
    const int wave = tid >> 5;
    const int hi   = lane >> 4;
    const int nn   = lane & 15;

    __shared__ __align__(16) _Float16 xsh[BT][STR];   // ~38 KB
    __shared__ __align__(16) _Float16 bshT[JC][STR];  // ~19 KB (transposed)
    __shared__ float red[BT];

    const float* Tk = T + (size_t)k * DIN * DIN;
    const int arow = wave * 16 + nn;       // A-fragment row (m = lane%16)

    float partial[8];
#pragma unroll
    for (int r = 0; r < 8; ++r) partial[r] = 0.f;

    for (int ip = 0; ip < DIN; ip += IP) {
        __syncthreads();                    // xsh readers of prev panel done
        // stage X panel: float4 global reads -> packed f16 (b64 LDS stores)
        for (int idx = tid; idx < BT * IP / 4; idx += 256) {
            const int r  = idx >> 5;            // / (IP/4)
            const int c4 = (idx & 31) << 2;     // % (IP/4) * 4
            const float4 vv = *(const float4*)&X[(size_t)(b0 + r) * DIN + ip + c4];
            v4h hv = { (_Float16)vv.x, (_Float16)vv.y,
                       (_Float16)vv.z, (_Float16)vv.w };
            *(v4h*)&xsh[r][c4] = hv;
        }
        for (int jc = 0; jc < DIN; jc += JC) {
            __syncthreads();                // bshT readers done
            // stage T chunk transposed: bshT[j][i] = Tk[ip+i][jc+j]
            for (int idx = tid; idx < IP * JC / 4; idx += 256) {
                const int r  = idx >> 4;        // i within panel
                const int c4 = (idx & 15) << 2; // j within chunk
                const float4 vv = *(const float4*)&Tk[(size_t)(ip + r) * DIN + jc + c4];
                bshT[c4 + 0][r] = (_Float16)vv.x;
                bshT[c4 + 1][r] = (_Float16)vv.y;
                bshT[c4 + 2][r] = (_Float16)vv.z;
                bshT[c4 + 3][r] = (_Float16)vv.w;
            }
            __syncthreads();
            // prefetch next T chunk into cache while we compute on this one
            {
                const int jcn = jc + JC;
                if (jcn < DIN) {
                    const int r = tid >> 1, h = (tid & 1) * 32;
                    __builtin_prefetch(&Tk[(size_t)(ip + r) * DIN + jcn + h], 0, 2);
                } else if (ip + IP < DIN) {
                    const int r = tid >> 1, h = (tid & 1) * 32;
                    __builtin_prefetch(&Tk[(size_t)(ip + IP + r) * DIN + h], 0, 2);
                }
            }
#pragma unroll
            for (int jt = 0; jt < JC; jt += 16) {
                CFrag C;
#pragma unroll
                for (int r = 0; r < 8; ++r) C.f[r] = 0.f;
#pragma unroll
                for (int is = 0; is < IP; is += 32) {
                    AFrag A, Bf;
                    A.g[0] = *(const v8h*)&xsh[arow][is + 16 * hi];
                    A.g[1] = *(const v8h*)&xsh[arow][is + 16 * hi + 8];
                    const _Float16* bp = &bshT[jt + nn][is + 16 * hi];
                    Bf.g[0] = *(const v8h*)bp;
                    Bf.g[1] = *(const v8h*)(bp + 8);
                    C.v = __builtin_amdgcn_wmma_f32_16x16x32_f16(
                              false, A.v, false, Bf.v, (short)0, C.v, false, false);
                }
                // fold P tile against Y (linear in C, so per-i-panel partial
                // folds sum to the full bilinear)
                const int ycol = jc + jt + nn;
#pragma unroll
                for (int r = 0; r < 8; ++r) {
                    const int m = r + 8 * hi;
                    partial[r] = fmaf(C.f[r],
                                      Y[(size_t)(b0 + wave * 16 + m) * DIN + ycol],
                                      partial[r]);
                }
            }
        }
    }

    // Deterministic reduction over the 16 lanes (n) of each hi-group.
#pragma unroll
    for (int r = 0; r < 8; ++r) {
        float p = partial[r];
#pragma unroll
        for (int msk = 8; msk >= 1; msk >>= 1)
            p += __shfl_xor(p, msk, 32);
        partial[r] = p;
    }
    __syncthreads();
    if (nn == 0) {
#pragma unroll
        for (int r = 0; r < 8; ++r)
            red[wave * 16 + r + 8 * hi] = partial[r];
    }
    __syncthreads();
    if (tid < BT) {
        const int b = b0 + tid;
        const float v = ACCin[(size_t)b * KDIM + k] + red[tid];
        OUT[(size_t)b * KDIM + k] = tanhf(v);
    }
}

// ---------------------------------------------------------------------------
extern "C" void kernel_launch(void* const* d_in, const int* in_sizes, int n_in,
                              void* d_out, int out_size, void* d_ws, size_t ws_size,
                              hipStream_t stream) {
    (void)in_sizes; (void)n_in; (void)out_size; (void)ws_size;

    const int*   subj_id = (const int*)  d_in[0];
    const float* subj_w  = (const float*)d_in[1];
    const int*   verb_id = (const int*)  d_in[2];
    const float* verb_w  = (const float*)d_in[3];
    const int*   obj_id  = (const int*)  d_in[4];
    const float* obj_w   = (const float*)d_in[5];
    const float* emb     = (const float*)d_in[6];
    const float* t1      = (const float*)d_in[7];
    const float* t2      = (const float*)d_in[8];
    const float* t3      = (const float*)d_in[9];
    const float* W1      = (const float*)d_in[10];
    const float* b1      = (const float*)d_in[11];
    const float* W2      = (const float*)d_in[12];
    const float* b2      = (const float*)d_in[13];
    const float* W3      = (const float*)d_in[14];
    const float* b3      = (const float*)d_in[15];

    float* ws   = (float*)d_ws;
    float* subj = ws;                                // B*D
    float* verb = subj + (size_t)BATCH * DDIM;       // B*D
    float* obj  = verb + (size_t)BATCH * DDIM;       // B*D
    float* acc1 = obj  + (size_t)BATCH * DDIM;       // B*K
    float* acc2 = acc1 + (size_t)BATCH * KDIM;       // B*K
    float* out  = (float*)d_out;                     // B*K

    // Stage 0: phrase embeddings
    phrase_embed_kernel<<<BATCH, 256, 0, stream>>>(subj_id, subj_w, emb, subj);
    phrase_embed_kernel<<<BATCH, 256, 0, stream>>>(verb_id, verb_w, emb, verb);
    phrase_embed_kernel<<<BATCH, 256, 0, stream>>>(obj_id,  obj_w,  emb, obj);

    // Stage 1/2 linear parts: acc = concat @ W^T + b
    linear_wmma_kernel<<<512, 256, 0, stream>>>(subj, verb, W1, b1, acc1, DDIM);
    linear_wmma_kernel<<<512, 256, 0, stream>>>(verb, obj,  W2, b2, acc2, DDIM);

    // Stage 1/2 bilinear + tanh (in place on acc1/acc2 -> r1/r2)
    dim3 bg(BATCH / 128, KDIM);
    bilinear_tanh_kernel<DDIM><<<bg, 256, 0, stream>>>(t1, subj, verb, acc1, acc1);
    bilinear_tanh_kernel<DDIM><<<bg, 256, 0, stream>>>(t2, verb, obj,  acc2, acc2);

    // Stage 3: linear part into d_out, then bilinear(t3, r1, r2) + tanh
    linear_wmma_kernel<<<512, 256, 0, stream>>>(acc1, acc2, W3, b3, out, KDIM);
    bilinear_tanh_kernel<KDIM><<<bg, 256, 0, stream>>>(t3, acc1, acc2, out, out);
}
